// Model_39676907883891
// MI455X (gfx1250) — matrix operand order repaired and checked by reading, not compiled, not run on hardware
//
#include <hip/hip_runtime.h>
#include <hip/hip_bf16.h>
#include <stdint.h>

typedef __attribute__((ext_vector_type(16))) _Float16 v16h;
typedef __attribute__((ext_vector_type(8)))  _Float16 v8h;
typedef __attribute__((ext_vector_type(2)))  _Float16 v2h;
typedef __attribute__((ext_vector_type(8)))  float    v8f;

#define BATCH 32
#define SEQ   2048
#define HDIM  64
#define ROWS_PER_BLOCK 128   // 8 waves x 16 rows
#define COLS_PER_TILE  64    // K columns staged in LDS per iteration
#define KS_STRIDE 72         // padded (halves) -> conflict-free b128 loads
#define QS_STRIDE 72
#define HALF_N 67108864u     // (B/2)*S*S = 2^26: threefry pairs (i, i+HALF_N)
                             // = same (row,col) in batch b and b+16

// Exact JAX threefry2x32 with key = (0, 42) (jax.random.key(42)).
// One call yields BOTH output words -> two dropout mask elements.
__device__ __forceinline__ uint32_t rotl32(uint32_t v, int r) {
  return (v << r) | (v >> (32 - r));
}
__device__ __forceinline__ void threefry2x32(uint32_t c0, uint32_t c1,
                                             uint32_t& o0, uint32_t& o1) {
  const uint32_t k0 = 0u, k1 = 42u;
  const uint32_t k2 = k0 ^ k1 ^ 0x1BD11BDAu;
  uint32_t x0 = c0 + k0, x1 = c1 + k1;
#define TF_R(r) { x0 += x1; x1 = rotl32(x1, r); x1 ^= x0; }
  TF_R(13) TF_R(15) TF_R(26) TF_R(6)
  x0 += k1; x1 += k2 + 1u;
  TF_R(17) TF_R(29) TF_R(16) TF_R(24)
  x0 += k2; x1 += k0 + 2u;
  TF_R(13) TF_R(15) TF_R(26) TF_R(6)
  x0 += k0; x1 += k1 + 3u;
  TF_R(17) TF_R(29) TF_R(16) TF_R(24)
  x0 += k1; x1 += k2 + 4u;
  TF_R(13) TF_R(15) TF_R(26) TF_R(6)
  x0 += k2; x1 += k0 + 5u;
#undef TF_R
  o0 = x0; o1 = x1;
}

extern "C" __global__ __launch_bounds__(256, 1)
void attn_probs_dropout(const float* __restrict__ Q,
                        const float* __restrict__ Kmat,
                        const float* __restrict__ scale_p,
                        const float* __restrict__ drop_p,
                        float* __restrict__ out) {
  // pb = 0 -> batch b0 (< 16), pb = 1 -> batch b0 + 16 (threefry word pair)
  __shared__ _Float16 Qs[2][8][16 * QS_STRIDE];
  __shared__ _Float16 Ks[2][COLS_PER_TILE * KS_STRIDE];

  const int tid  = threadIdx.x;
  const int lane = tid & 31;
  const int wave = tid >> 5;
  const int hh   = lane >> 4;      // wave half: 0 = lanes 0-15, 1 = lanes 16-31
  const int lcol = lane & 15;
  const int b0   = blockIdx.y;               // 0..15
  const int qBase = blockIdx.x * ROWS_PER_BLOCK + wave * 16;

  const float scale    = scale_p[0];
  const float keep_p   = 1.0f - drop_p[0];
  const float inv_keep = 1.0f / keep_p;

  const float* ksrc[2] = { Kmat + (size_t)b0 * SEQ * HDIM,
                           Kmat + (size_t)(b0 + BATCH / 2) * SEQ * HDIM };

  // ---- stage this wave's 16x64 Q tiles (both batches) into LDS as f16 ----
  #pragma unroll
  for (int pb = 0; pb < 2; ++pb) {
    const float* qsrc =
        Q + ((size_t)(b0 + pb * (BATCH / 2)) * SEQ + (size_t)qBase) * HDIM;
    #pragma unroll
    for (int ii = 0; ii < 16; ++ii) {
      int idx = (ii * 32 + lane) * 2;
      int r = idx >> 6, d = idx & 63;
      v2h h; h[0] = (_Float16)qsrc[r * HDIM + d]; h[1] = (_Float16)qsrc[r * HDIM + d + 1];
      *(v2h*)&Qs[pb][wave][r * QS_STRIDE + d] = h;
    }
  }
  __syncthreads();

  // ---- assemble A fragments (16x32 f16, contiguous k = 16*half + j, matching B), kept in VGPRs ----
  v16h a0[2], a1[2];
  #pragma unroll
  for (int pb = 0; pb < 2; ++pb) {
    #pragma unroll
    for (int j = 0; j < 16; ++j) {
      int k = hh * 16 + j;
      a0[pb][j] = Qs[pb][wave][lcol * QS_STRIDE + k];        // K = 0..31
      a1[pb][j] = Qs[pb][wave][lcol * QS_STRIDE + 32 + k];   // K = 32..63
    }
  }

  float m[2][8], l[2][8];
  #pragma unroll
  for (int pb = 0; pb < 2; ++pb)
    #pragma unroll
    for (int r = 0; r < 8; ++r) { m[pb][r] = -3.0e38f; l[pb][r] = 0.0f; }

  // =================== pass 1: row max + sum of exp ===================
  for (int t = 0; t < SEQ / COLS_PER_TILE; ++t) {
    __syncthreads();
    #pragma unroll
    for (int pb = 0; pb < 2; ++pb) {
      #pragma unroll
      for (int ii = 0; ii < (COLS_PER_TILE * HDIM) / 512; ++ii) {
        int idx = (ii * 256 + tid) * 2;
        int c = idx >> 6, d = idx & 63;
        const float* p = &ksrc[pb][(size_t)(t * COLS_PER_TILE + c) * HDIM + d];
        v2h h; h[0] = (_Float16)p[0]; h[1] = (_Float16)p[1];
        *(v2h*)&Ks[pb][c * KS_STRIDE + d] = h;
      }
    }
    __syncthreads();
    // prefetch next K tile while this one is consumed (global_prefetch_b8)
    if (t + 1 < SEQ / COLS_PER_TILE) {
      __builtin_prefetch(&ksrc[0][(size_t)((t + 1) * COLS_PER_TILE) * HDIM + tid * 16], 0, 1);
      __builtin_prefetch(&ksrc[1][(size_t)((t + 1) * COLS_PER_TILE) * HDIM + tid * 16], 0, 1);
    }
    #pragma unroll
    for (int nt = 0; nt < COLS_PER_TILE / 16; ++nt) {
      int col = nt * 16 + lcol;
      #pragma unroll
      for (int pb = 0; pb < 2; ++pb) {
        v8h b0lo = *(const v8h*)&Ks[pb][col * KS_STRIDE + hh * 16];
        v8h b0hi = *(const v8h*)&Ks[pb][col * KS_STRIDE + hh * 16 + 8];
        v8h b1lo = *(const v8h*)&Ks[pb][col * KS_STRIDE + 32 + hh * 16];
        v8h b1hi = *(const v8h*)&Ks[pb][col * KS_STRIDE + 32 + hh * 16 + 8];
        v16h bf0 = __builtin_shufflevector(b0lo, b0hi, 0,1,2,3,4,5,6,7,8,9,10,11,12,13,14,15);
        v16h bf1 = __builtin_shufflevector(b1lo, b1hi, 0,1,2,3,4,5,6,7,8,9,10,11,12,13,14,15);
        v8f acc = {};
        acc = __builtin_amdgcn_wmma_f32_16x16x32_f16(false, a0[pb], false, bf0, (short)0, acc, false, false);
        acc = __builtin_amdgcn_wmma_f32_16x16x32_f16(false, a1[pb], false, bf1, (short)0, acc, false, false);
        #pragma unroll
        for (int r = 0; r < 8; ++r) {
          float s  = acc[r] * scale;
          float mn = fmaxf(m[pb][r], s);
          l[pb][r] = l[pb][r] * __expf(m[pb][r] - mn) + __expf(s - mn);
          m[pb][r] = mn;
        }
      }
    }
  }

  // ---- merge (m,l) across the 16 lanes holding the same rows ----
  #pragma unroll
  for (int off = 1; off < 16; off <<= 1) {
    #pragma unroll
    for (int pb = 0; pb < 2; ++pb) {
      #pragma unroll
      for (int r = 0; r < 8; ++r) {
        float mo = __shfl_xor(m[pb][r], off, 32);
        float lo = __shfl_xor(l[pb][r], off, 32);
        float mn = fmaxf(m[pb][r], mo);
        l[pb][r] = l[pb][r] * __expf(m[pb][r] - mn) + lo * __expf(mo - mn);
        m[pb][r] = mn;
      }
    }
  }
  float invl[2][8];
  #pragma unroll
  for (int pb = 0; pb < 2; ++pb)
    #pragma unroll
    for (int r = 0; r < 8; ++r) invl[pb][r] = 1.0f / l[pb][r];

  // ===== pass 2: recompute, normalize, shared-threefry dropout, store =====
  for (int t = 0; t < SEQ / COLS_PER_TILE; ++t) {
    __syncthreads();
    #pragma unroll
    for (int pb = 0; pb < 2; ++pb) {
      #pragma unroll
      for (int ii = 0; ii < (COLS_PER_TILE * HDIM) / 512; ++ii) {
        int idx = (ii * 256 + tid) * 2;
        int c = idx >> 6, d = idx & 63;
        const float* p = &ksrc[pb][(size_t)(t * COLS_PER_TILE + c) * HDIM + d];
        v2h h; h[0] = (_Float16)p[0]; h[1] = (_Float16)p[1];
        *(v2h*)&Ks[pb][c * KS_STRIDE + d] = h;
      }
    }
    __syncthreads();
    if (t + 1 < SEQ / COLS_PER_TILE) {
      __builtin_prefetch(&ksrc[0][(size_t)((t + 1) * COLS_PER_TILE) * HDIM + tid * 16], 0, 1);
      __builtin_prefetch(&ksrc[1][(size_t)((t + 1) * COLS_PER_TILE) * HDIM + tid * 16], 0, 1);
    }
    #pragma unroll
    for (int nt = 0; nt < COLS_PER_TILE / 16; ++nt) {
      int col = nt * 16 + lcol;
      v8f acc[2];
      #pragma unroll
      for (int pb = 0; pb < 2; ++pb) {
        v8h b0lo = *(const v8h*)&Ks[pb][col * KS_STRIDE + hh * 16];
        v8h b0hi = *(const v8h*)&Ks[pb][col * KS_STRIDE + hh * 16 + 8];
        v8h b1lo = *(const v8h*)&Ks[pb][col * KS_STRIDE + 32 + hh * 16];
        v8h b1hi = *(const v8h*)&Ks[pb][col * KS_STRIDE + 32 + hh * 16 + 8];
        v16h bf0 = __builtin_shufflevector(b0lo, b0hi, 0,1,2,3,4,5,6,7,8,9,10,11,12,13,14,15);
        v16h bf1 = __builtin_shufflevector(b1lo, b1hi, 0,1,2,3,4,5,6,7,8,9,10,11,12,13,14,15);
        v8f a = {};
        a = __builtin_amdgcn_wmma_f32_16x16x32_f16(false, a0[pb], false, bf0, (short)0, a, false, false);
        a = __builtin_amdgcn_wmma_f32_16x16x32_f16(false, a1[pb], false, bf1, (short)0, a, false, false);
        acc[pb] = a;
      }
      int colg = t * COLS_PER_TILE + nt * 16 + lcol;
      #pragma unroll
      for (int r = 0; r < 8; ++r) {
        int rowg = qBase + r + 8 * hh;
        float p0 = __expf(acc[0][r] * scale - m[0][r]) * invl[0][r];
        float p1 = __expf(acc[1][r] * scale - m[1][r]) * invl[1][r];
        // counter i < 2^26 pairs with i + 2^26: same (row,col), batches b0 / b0+16
        uint32_t i0 = (uint32_t)(b0 * SEQ + rowg) * (uint32_t)SEQ + (uint32_t)colg;
        uint32_t w0, w1;
        threefry2x32(i0, i0 + HALF_N, w0, w1);
        float u0 = __uint_as_float((w0 >> 9) | 0x3f800000u) - 1.0f;
        float u1 = __uint_as_float((w1 >> 9) | 0x3f800000u) - 1.0f;
        float v0 = (u0 < keep_p) ? p0 * inv_keep : 0.0f;
        float v1 = (u1 < keep_p) ? p1 * inv_keep : 0.0f;
        __builtin_nontemporal_store(v0, &out[(size_t)i0]);
        __builtin_nontemporal_store(v1, &out[(size_t)i0 + HALF_N]);
      }
    }
  }
}

extern "C" void kernel_launch(void* const* d_in, const int* in_sizes, int n_in,
                              void* d_out, int out_size, void* d_ws, size_t ws_size,
                              hipStream_t stream) {
  const float* Q     = (const float*)d_in[0];
  const float* K     = (const float*)d_in[1];
  const float* scale = (const float*)d_in[2];
  const float* dropp = (const float*)d_in[3];
  // d_in[4] = value: dead code in the reference
  float* out = (float*)d_out;
  dim3 grid(SEQ / ROWS_PER_BLOCK, BATCH / 2);
  attn_probs_dropout<<<grid, dim3(256), 0, stream>>>(Q, K, scale, dropp, out);
  (void)in_sizes; (void)n_in; (void)out_size; (void)d_ws; (void)ws_size;
}
